// FullTensorSquareSelfInteraction_24902220383051
// MI455X (gfx1250) — matrix operand order repaired and checked by reading, not compiled, not run on hardware
//
#include <hip/hip_runtime.h>
#include <math.h>

// ---------------- problem constants (from reference) ----------------
// M0=32, M1=16, N=32768
// D0=696, D1O=528, D1E=120, D2E=136, NIRR=1480
// outputs: o0(128) | o1o 64x3 | o1e 32x3 | o2e 32x5  => 576 floats/row
#define NROWS   32768
#define XCOLS   80
#define OUTC    576

// padded K dims (multiples of 32 for wmma 16x16x32)
#define K0P     704     // D0 696 -> 704
#define K1OP    544     // 528 -> 544
#define K1EP    128     // 120 -> 128
#define K2EP    160     // 136 -> 160

// f16 workspace layout (element offsets)
#define OFF_W1T   ((size_t)0)              // [704][704]
#define OFF_W2T   ((size_t)495616)         // [1488][704]
#define OFF_W0ET  ((size_t)1543168)        // [128][704]
#define OFF_W1OT  ((size_t)1633280)        // [64][544]
#define OFF_W1ET  ((size_t)1668096)        // [32][128]
#define OFF_W2ET  ((size_t)1672192)        // [32][160]
// total = 1,677,312 halves = 3,354,624 bytes

typedef _Float16 h16;
typedef __attribute__((ext_vector_type(8)))  _Float16 v8h;
typedef __attribute__((ext_vector_type(16))) _Float16 v16h;
typedef __attribute__((ext_vector_type(8)))  float    v8f;

__device__ __forceinline__ float silu_f(float x) {
    return x * (1.0f / (1.0f + __expf(-x)));
}

// One 16x16 output tile of D = A(16xK) * B(Kx16), K = 32*ktiles.
// A: row-major f16 in LDS, pitch apitch (halves). Lane l<16: M=l, K-halves
// [kt*32+0..7] and [+16..23]; lane>=16: M=l-16, [+8..15] and [+24..31].
// B: column-major f16 (Bt[n][k], pitch bpitch). Lane col = l&15;
// lanes 0-15 hold K 0..7 and 16..23, lanes 16-31 hold K 8..15 and 24..31.
__device__ __forceinline__ v8f gemm_tile(const h16* As, int apitch,
                                         const h16* Bt, int bpitch,
                                         int ncol0, int ktiles, int lane) {
    const int mrow = lane & 15;
    const int sel  = lane >> 4;
    const h16* ap = As + mrow * apitch + sel * 8;
    const h16* bp = Bt + (size_t)(ncol0 + mrow) * bpitch + sel * 8;
    v8f acc = {};
    for (int kt = 0; kt < ktiles; ++kt) {
        v8h alo = *(const v8h*)(ap);
        v8h ahi = *(const v8h*)(ap + 16);
        v8h blo = *(const v8h*)(bp);
        v8h bhi = *(const v8h*)(bp + 16);
        v16h A = __builtin_shufflevector(alo, ahi, 0,1,2,3,4,5,6,7,8,9,10,11,12,13,14,15);
        v16h B = __builtin_shufflevector(blo, bhi, 0,1,2,3,4,5,6,7,8,9,10,11,12,13,14,15);
        acc = __builtin_amdgcn_wmma_f32_16x16x32_f16(false, A, false, B,
                                                     (short)0, acc, false, false);
        ap += 32;
        bp += 32;
    }
    return acc;
}

// Pre-pass: transpose + pad + pre-scale weights to f16 column-major Bt[n][k].
__global__ void convert_wt(const float* __restrict__ src, h16* __restrict__ dst,
                           int K, int Nn, int Kpad, int Npad, float scale) {
    int total = Kpad * Npad;
    for (int idx = blockIdx.x * blockDim.x + threadIdx.x; idx < total;
         idx += gridDim.x * blockDim.x) {
        int n = idx / Kpad;
        int k = idx - n * Kpad;
        float v = 0.0f;
        if (n < Nn && k < K) v = src[(size_t)k * Nn + n] * scale;
        dst[(size_t)n * Kpad + k] = (h16)v;
    }
}

__global__ __launch_bounds__(256)
void fused_forward(const float* __restrict__ x, const h16* __restrict__ ws,
                   float* __restrict__ out) {
    // LDS: ~138.5 KB total
    __shared__ __align__(16) h16 sA0[16 * K0P];        // f0 (later gated)
    __shared__ __align__(16) h16 sA1o[3][16 * K1OP];   // f1o per channel
    __shared__ __align__(16) h16 sA1e[3][16 * K1EP];   // cross per channel
    __shared__ __align__(16) h16 sA2e[5][16 * K2EP];   // quad per component
    __shared__ __align__(16) h16 sH[16 * K0P];         // silu(f0 @ W1s)
    __shared__ __align__(16) float xs[16 * XCOLS];
    __shared__ unsigned char t32i[528], t32j[528];
    __shared__ unsigned char t16i[136], t16j[136];
    __shared__ unsigned char tki[120],  tkj[120];

    const int tid  = threadIdx.x;
    const int lane = tid & 31;
    const int wave = tid >> 5;
    const int row0 = blockIdx.x * 16;

    // ---- stage 0: x tile + triu index tables ----
    for (int i = tid; i < 16 * XCOLS; i += 256)
        xs[i] = x[(size_t)row0 * XCOLS + i];
    for (int p = tid; p < 528; p += 256) {       // triu(32)
        int i = 0, base = 0;
        while (base + (32 - i) <= p) { base += 32 - i; ++i; }
        t32i[p] = (unsigned char)i;
        t32j[p] = (unsigned char)(i + (p - base));
    }
    for (int p = tid; p < 136; p += 256) {       // triu(16)
        int i = 0, base = 0;
        while (base + (16 - i) <= p) { base += 16 - i; ++i; }
        t16i[p] = (unsigned char)i;
        t16j[p] = (unsigned char)(i + (p - base));
    }
    for (int p = tid; p < 120; p += 256) {       // triu(16, k=1)
        int i = 0, base = 0;
        while (base + (15 - i) <= p) { base += 15 - i; ++i; }
        tki[p] = (unsigned char)i;
        tkj[p] = (unsigned char)(i + 1 + (p - base));
    }
    __syncthreads();

    // ---- stage 1: equivariant features (f16 A-operands, zero-padded) ----
    {
        const int r  = tid & 15;
        const int wk = tid >> 4;
        const float* xr = &xs[r * XCOLS];
        const float i_r2 = 0.70710678118f;
        const float i_r3 = 0.57735026919f;
        const float i_r6 = 0.40824829046f;
        #define SS(i)    xr[i]
        #define VV(j, c) xr[32 + (j) * 3 + (c)]
        for (int p = wk; p < K0P; p += 16) {                 // f0
            float val = 0.0f;
            if (p < 32) val = SS(p);
            else if (p < 560) { int q = p - 32; val = SS(t32i[q]) * SS(t32j[q]); }
            else if (p < 696) {
                int q = p - 560; int i = t16i[q], j = t16j[q];
                val = (VV(i,0)*VV(j,0) + VV(i,1)*VV(j,1) + VV(i,2)*VV(j,2)) * i_r3;
            }
            sA0[r * K0P + p] = (h16)val;
        }
        for (int p = wk; p < K1OP; p += 16) {                // f1o = [v, s⊗v]
            for (int c = 0; c < 3; ++c) {
                float val = 0.0f;
                if (p < 16) val = VV(p, c);
                else if (p < 528) { int q = p - 16; val = SS(q >> 4) * VV(q & 15, c); }
                sA1o[c][r * K1OP + p] = (h16)val;
            }
        }
        for (int p = wk; p < K1EP; p += 16) {                // f1e = cross/√2
            float c0 = 0.0f, c1 = 0.0f, c2 = 0.0f;
            if (p < 120) {
                int i = tki[p], j = tkj[p];
                float ax=VV(i,0), ay=VV(i,1), az=VV(i,2);
                float bx=VV(j,0), by=VV(j,1), bz=VV(j,2);
                c0 = (ay*bz - az*by) * i_r2;
                c1 = (az*bx - ax*bz) * i_r2;
                c2 = (ax*by - ay*bx) * i_r2;
            }
            sA1e[0][r * K1EP + p] = (h16)c0;
            sA1e[1][r * K1EP + p] = (h16)c1;
            sA1e[2][r * K1EP + p] = (h16)c2;
        }
        for (int p = wk; p < K2EP; p += 16) {                // f2e = quad
            float q0=0,q1=0,q2=0,q3=0,q4=0;
            if (p < 136) {
                int i = t16i[p], j = t16j[p];
                float ax=VV(i,0), ay=VV(i,1), az=VV(i,2);
                float bx=VV(j,0), by=VV(j,1), bz=VV(j,2);
                q0 = (ax*by + ay*bx) * i_r2;
                q1 = (ay*bz + az*by) * i_r2;
                q2 = (2.0f*az*bz - ax*bx - ay*by) * i_r6;
                q3 = (ax*bz + az*bx) * i_r2;
                q4 = (ax*bx - ay*by) * i_r2;
            }
            sA2e[0][r * K2EP + p] = (h16)q0;
            sA2e[1][r * K2EP + p] = (h16)q1;
            sA2e[2][r * K2EP + p] = (h16)q2;
            sA2e[3][r * K2EP + p] = (h16)q3;
            sA2e[4][r * K2EP + p] = (h16)q4;
        }
        #undef SS
        #undef VV
    }
    __syncthreads();

    const int mrow = lane & 15;
    const int sel  = lane >> 4;

    // ---- stage 2: h = silu(f0 @ W1s)  (44 N-tiles over 8 waves) ----
    for (int nt = wave; nt < 44; nt += 8) {
        v8f acc = gemm_tile(sA0, K0P, ws + OFF_W1T, K0P, nt * 16, 22, lane);
        for (int rr = 0; rr < 8; ++rr) {
            int m = rr + sel * 8;
            sH[m * K0P + nt * 16 + mrow] = (h16)silu_f(acc[rr]);
        }
    }
    __syncthreads();

    // ---- stage 3: g = silu(h @ W2s), fused gating f *= g  (93 N-tiles) ----
    for (int nt = wave; nt < 93; nt += 8) {
        v8f acc = gemm_tile(sH, K0P, ws + OFF_W2T, K0P, nt * 16, 22, lane);
        int n = nt * 16 + mrow;
        for (int rr = 0; rr < 8; ++rr) {
            int m = rr + sel * 8;
            float g = silu_f(acc[rr]);
            if (n < 696) {
                sA0[m * K0P + n] = (h16)((float)sA0[m * K0P + n] * g);
            } else if (n < 1224) {
                int q = n - 696;
                for (int c = 0; c < 3; ++c)
                    sA1o[c][m * K1OP + q] = (h16)((float)sA1o[c][m * K1OP + q] * g);
            } else if (n < 1344) {
                int q = n - 1224;
                for (int c = 0; c < 3; ++c)
                    sA1e[c][m * K1EP + q] = (h16)((float)sA1e[c][m * K1EP + q] * g);
            } else if (n < 1480) {
                int q = n - 1344;
                for (int c = 0; c < 5; ++c)
                    sA2e[c][m * K2EP + q] = (h16)((float)sA2e[c][m * K2EP + q] * g);
            }
        }
    }
    __syncthreads();

    // ---- stage 4: output GEMMs (36 tile-jobs over 8 waves) ----
    // jobs 0-7:   o0   8 tiles, K=704,  cols 0..127  stride 1
    // jobs 8-19:  o1o  3ch x 4 tiles, K=544, cols 128+o*3+c
    // jobs 20-25: o1e  3ch x 2 tiles, K=128, cols 320+o*3+c
    // jobs 26-35: o2e  5ch x 2 tiles, K=160, cols 416+o*5+c
    float* outp = out + (size_t)row0 * OUTC;
    for (int job = wave; job < 36; job += 8) {
        const h16* A; const h16* B;
        int apitch, ktiles, nt, colbase, cstride;
        if (job < 8) {
            A = sA0; apitch = K0P; ktiles = 22;
            B = ws + OFF_W0ET; nt = job; colbase = 0; cstride = 1;
        } else if (job < 20) {
            int t = job - 8; int c = t >> 2; nt = t & 3;
            A = &sA1o[c][0]; apitch = K1OP; ktiles = 17;
            B = ws + OFF_W1OT; colbase = 128 + c; cstride = 3;
        } else if (job < 26) {
            int t = job - 20; int c = t >> 1; nt = t & 1;
            A = &sA1e[c][0]; apitch = K1EP; ktiles = 4;
            B = ws + OFF_W1ET; colbase = 320 + c; cstride = 3;
        } else {
            int t = job - 26; int c = t >> 1; nt = t & 1;
            A = &sA2e[c][0]; apitch = K2EP; ktiles = 5;
            B = ws + OFF_W2ET; colbase = 416 + c; cstride = 5;
        }
        v8f acc = gemm_tile(A, apitch, B, apitch, nt * 16, ktiles, lane);
        int col = colbase + (nt * 16 + mrow) * cstride;
        for (int rr = 0; rr < 8; ++rr)
            outp[(size_t)(rr + sel * 8) * OUTC + col] = acc[rr];
    }
}

extern "C" void kernel_launch(void* const* d_in, const int* in_sizes, int n_in,
                              void* d_out, int out_size, void* d_ws, size_t ws_size,
                              hipStream_t stream) {
    const float* x   = (const float*)d_in[0];
    const float* W1  = (const float*)d_in[1];
    const float* W2  = (const float*)d_in[2];
    const float* w0e = (const float*)d_in[3];
    const float* w1o = (const float*)d_in[4];
    const float* w1e = (const float*)d_in[5];
    const float* w2e = (const float*)d_in[6];
    float* out = (float*)d_out;
    h16* wsh = (h16*)d_ws;

    const float sD0 = 1.0f / sqrtf(696.0f);
    const float s1o = 1.0f / sqrtf(528.0f);
    const float s1e = 1.0f / sqrtf(120.0f);
    const float s2e = 1.0f / sqrtf(136.0f);

    // Pre-scale + transpose + pad all weights to f16 once (L2-resident after).
    convert_wt<<<512, 256, 0, stream>>>(W1,  wsh + OFF_W1T,  696, 696,  K0P,  704,  sD0);
    convert_wt<<<512, 256, 0, stream>>>(W2,  wsh + OFF_W2T,  696, 1480, K0P,  1488, sD0);
    convert_wt<<<512, 256, 0, stream>>>(w0e, wsh + OFF_W0ET, 696, 128,  K0P,  128,  sD0);
    convert_wt<<<512, 256, 0, stream>>>(w1o, wsh + OFF_W1OT, 528, 64,   K1OP, 64,   s1o);
    convert_wt<<<512, 256, 0, stream>>>(w1e, wsh + OFF_W1ET, 120, 32,   K1EP, 32,   s1e);
    convert_wt<<<512, 256, 0, stream>>>(w2e, wsh + OFF_W2ET, 136, 32,   K2EP, 32,   s2e);

    fused_forward<<<NROWS / 16, 256, 0, stream>>>(x, wsh, out);
}
